// JTNNEncoder_12232066859185
// MI455X (gfx1250) — compile-verified, matrix operand order checked by reading, not run on hardware
//
#include <hip/hip_runtime.h>
#include <hip/hip_bf16.h>
#include <math.h>

// ---------------- problem constants (from the reference) ----------------
#define HH      256          // hidden size
#define HH2     512          // 2*H
#define BB      4096         // batch
#define KK      8            // max edges per BFS level
#define BKROWS  (BB * KK)    // rows per step = 32768
#define TSTEPS  8            // BFS schedule length
#define MAXNB   8
#define EDGES   30           // 2*(16-1)
#define HSLOTS  (1 + BB * EDGES)   // 122881 message slots

typedef __attribute__((ext_vector_type(16))) _Float16 v16h;
typedef __attribute__((ext_vector_type(8)))  float    v8f;

// ---------------- WMMA helpers (CDNA5 16x16x32 f16 -> f32) ----------------
__device__ __forceinline__ v8f wmma_f32_16x16x32(v16h a, v16h b, v8f c) {
  // 8 args: (neg_a, A, neg_b, B, c_mod, C, reuse_a, reuse_b)
  return __builtin_amdgcn_wmma_f32_16x16x32_f16(false, a, false, b, (short)0, c,
                                                false, false);
}

// ---- fragment swizzle (ISA 7.12.2, 16-bit operands) ----
// Within a 16(M)x32(K) [or 32(K)x16(N)] tile, element (m, kk) lives at
// lane = m + 16*half, slot j, where:
//   kk in [0,8)   -> half=0, j=kk
//   kk in [8,16)  -> half=1, j=kk-8
//   kk in [16,24) -> half=0, j=kk-8
//   kk in [24,32) -> half=1, j=kk-16
// A tile is stored as 32 lanes x 16 contiguous halfs = 512 halfs (1 KB).
__device__ __forceinline__ int swz_off(int m, int kk) {
  const int t    = kk >> 3;
  const int half = t & 1;
  const int j    = kk - (((t + 1) >> 1) << 3);
  return ((m + (half << 4)) << 4) + j;
}

// A fragment from a swizzled LDS buffer: one 32-byte contiguous load per lane.
__device__ __forceinline__ v16h frag_sw(const _Float16* __restrict__ buf,
                                        int tile, int lane) {
  return *(const v16h*)(buf + (((tile << 5) + lane) << 4));
}

// ---------------- fast activations (hardware transcendentals) ----------------
__device__ __forceinline__ float fast_rcp(float x) {
#if __has_builtin(__builtin_amdgcn_rcpf)
  return __builtin_amdgcn_rcpf(x);
#else
  return 1.0f / x;
#endif
}
__device__ __forceinline__ float fast_sigmoid(float x) {
  return fast_rcp(1.0f + __expf(-x));
}
__device__ __forceinline__ float fast_tanh(float x) {
#if __has_builtin(__builtin_amdgcn_tanhf)
  return __builtin_amdgcn_tanhf(x);
#else
  return 2.0f * fast_sigmoid(2.0f * x) - 1.0f;
#endif
}

// ---------------- weight conversion f32 -> f16, pre-swizzled ----------------
// src: row-major (K x N) f32. dst: tiles of 32(K)x16(N) in fragment order,
// tile id = ntile * (K/32) + ktile, 512 halfs per tile.
__global__ void w_swizzle_kernel(const float* __restrict__ src,
                                 _Float16* __restrict__ dst,
                                 int Kdim, int Ndim) {
  const int o = blockIdx.x * blockDim.x + threadIdx.x;
  if (o >= Kdim * Ndim) return;
  const int tile   = o >> 9;
  const int lane   = (o >> 4) & 31;
  const int j      = o & 15;
  const int Ktiles = Kdim >> 5;
  const int ktile  = tile % Ktiles;
  const int ntile  = tile / Ktiles;
  const int k = (ktile << 5) + (((lane >> 4) & 1) << 3) + j + ((j >= 8) ? 8 : 0);
  const int n = (ntile << 4) + (lane & 15);
  dst[o] = (_Float16)src[k * Ndim + n];
}

// B fragment from swizzled global weights: one 32-byte coalesced load.
__device__ __forceinline__ v16h frag_w(const _Float16* __restrict__ W,
                                       int Ktiles, int ktile, int ntile,
                                       int lane) {
  return *(const v16h*)(W + ((((size_t)(ntile * Ktiles + ktile) << 5) + lane) << 4));
}

// ---------------- one BFS step of the tree-GRU ----------------
// Block: 16 rows, 256 threads (8 waves). Wave w owns columns [32w, 32w+32).
__global__ __launch_bounds__(256) void gru_step_kernel(
    const float*    __restrict__ emb,
    const _Float16* __restrict__ Wz16, const float* __restrict__ Wz_b,
    const _Float16* __restrict__ Wr16,
    const _Float16* __restrict__ Ur16, const float* __restrict__ Ur_b,
    const _Float16* __restrict__ Wh16, const float* __restrict__ Wh_b,
    float*          __restrict__ h,
    const int*      __restrict__ wid,  const int* __restrict__ nei,
    const int*      __restrict__ mid,  const float* __restrict__ mask) {
  // swizzled A operands
  __shared__ __align__(32) _Float16 sA[16 * HH2];   // [x | sum_h] (16 K-tiles)
  __shared__ __align__(32) _Float16 sHn[128 * HH];  // h_nei (8 M-tiles x 8 K-tiles)
  // linear staging
  __shared__ float    sSumH[16 * HH];
  __shared__ float    sSumG[16 * HH];
  __shared__ _Float16 sZ[16 * HH];
  __shared__ _Float16 sXWr[16 * HH];

  const int tid  = threadIdx.x;
  const int lane = tid & 31;
  const int wv   = tid >> 5;
  const int row0 = blockIdx.x * 16;
  const int nt0  = wv * 2;                 // wave's first N-tile
  const int n0   = wv * 32;

  // ---- Phase 1: gather x = emb[wid] and the 8 neighbor states ----
  for (int i = tid; i < 16 * HH; i += 256) {
    const int r = i >> 8, c = i & (HH - 1);
    const int w = wid[row0 + r];
    sA[((c >> 5) << 9) + swz_off(r, c & 31)] = (_Float16)emb[w * HH + c];
  }
  for (int i = tid; i < 128 * HH; i += 256) {
    const int s = i >> 8, c = i & (HH - 1);
    const int hidx = nei[(row0 + (s >> 3)) * MAXNB + (s & 7)];
    sHn[((((s >> 4) << 3) + (c >> 5)) << 9) + swz_off(s & 15, c & 31)] =
        (_Float16)h[(size_t)hidx * HH + c];
  }
  __syncthreads();
  for (int i = tid; i < 16 * HH; i += 256) {
    const int r = i >> 8, c = i & (HH - 1);
    const int ct = c >> 5, ck = c & 31;
    float s = 0.0f;
#pragma unroll
    for (int nb = 0; nb < MAXNB; ++nb) {
      const int sub = r * 8 + nb;
      s += (float)sHn[((((sub >> 4) << 3) + ct) << 9) + swz_off(sub & 15, ck)];
    }
    sSumH[i] = s;
    sA[((8 + ct) << 9) + swz_off(r, ck)] = (_Float16)s;   // K offset +256
  }
  __syncthreads();

  // ---- Phase 2a: z = sigmoid([x|sum_h] @ Wz + b) ----
  {
    v8f acc0 = {}, acc1 = {};
#pragma unroll
    for (int kt = 0; kt < 16; ++kt) {
      v16h a  = frag_sw(sA, kt, lane);
      v16h b0 = frag_w(Wz16, 16, kt, nt0, lane);
      v16h b1 = frag_w(Wz16, 16, kt, nt0 + 1, lane);
      acc0 = wmma_f32_16x16x32(a, b0, acc0);
      acc1 = wmma_f32_16x16x32(a, b1, acc1);
    }
    const int nA = n0 + (lane & 15), nB = nA + 16;
    const float bA = Wz_b[nA], bB = Wz_b[nB];
#pragma unroll
    for (int v = 0; v < 8; ++v) {
      const int m = v + ((lane >> 4) << 3);
      sZ[m * HH + nA] = (_Float16)fast_sigmoid(acc0[v] + bA);
      sZ[m * HH + nB] = (_Float16)fast_sigmoid(acc1[v] + bB);
    }
  }
  // ---- Phase 2b: xWr = x @ Wr ----
  {
    v8f acc0 = {}, acc1 = {};
#pragma unroll
    for (int kt = 0; kt < 8; ++kt) {
      v16h a  = frag_sw(sA, kt, lane);
      v16h b0 = frag_w(Wr16, 8, kt, nt0, lane);
      v16h b1 = frag_w(Wr16, 8, kt, nt0 + 1, lane);
      acc0 = wmma_f32_16x16x32(a, b0, acc0);
      acc1 = wmma_f32_16x16x32(a, b1, acc1);
    }
    const int nA = n0 + (lane & 15), nB = nA + 16;
#pragma unroll
    for (int v = 0; v < 8; ++v) {
      const int m = v + ((lane >> 4) << 3);
      sXWr[m * HH + nA] = (_Float16)acc0[v];
      sXWr[m * HH + nB] = (_Float16)acc1[v];
    }
  }

  // ---- Phase 3: r = sigmoid(xWr + h_nei @ Ur + b); sum_gated = sum r*h_nei
  // The D layout puts the 8 neighbors of one row into one lane's 8 VGPRs,
  // so the neighbor reduction is entirely in-lane.
  for (int mt = 0; mt < 8; ++mt) {
    v8f acc0 = {}, acc1 = {};
#pragma unroll
    for (int kt = 0; kt < 8; ++kt) {
      v16h a  = frag_sw(sHn, mt * 8 + kt, lane);
      v16h b0 = frag_w(Ur16, 8, kt, nt0, lane);
      v16h b1 = frag_w(Ur16, 8, kt, nt0 + 1, lane);
      acc0 = wmma_f32_16x16x32(a, b0, acc0);
      acc1 = wmma_f32_16x16x32(a, b1, acc1);
    }
    const int half = lane >> 4;
    const int row  = mt * 2 + half;
    const int nA = n0 + (lane & 15), nB = nA + 16;
    const int tA = nA >> 5, kA = nA & 31;
    const int tB = nB >> 5, kB = nB & 31;
    const float xwrA = (float)sXWr[row * HH + nA];
    const float xwrB = (float)sXWr[row * HH + nB];
    const float ubA = Ur_b[nA], ubB = Ur_b[nB];
    float sgA = 0.0f, sgB = 0.0f;
#pragma unroll
    for (int v = 0; v < 8; ++v) {
      const int sub = mt * 16 + v + half * 8;   // = row*8 + nb
      const int mm = sub & 15;
      const float hA = (float)sHn[((mt * 8 + tA) << 9) + swz_off(mm, kA)];
      const float hB = (float)sHn[((mt * 8 + tB) << 9) + swz_off(mm, kB)];
      sgA += fast_sigmoid(xwrA + acc0[v] + ubA) * hA;
      sgB += fast_sigmoid(xwrB + acc1[v] + ubB) * hB;
    }
    sSumG[row * HH + nA] = sgA;
    sSumG[row * HH + nB] = sgB;
  }
  __syncthreads();
  for (int i = tid; i < 16 * HH; i += 256) {
    const int r = i >> 8, c = i & (HH - 1);
    sA[((8 + (c >> 5)) << 9) + swz_off(r, c & 31)] = (_Float16)sSumG[i];
  }
  __syncthreads();

  // ---- Phase 4: pre_h = tanh([x|sum_gated] @ Wh + b); combine & scatter ----
  {
    v8f acc0 = {}, acc1 = {};
#pragma unroll
    for (int kt = 0; kt < 16; ++kt) {
      v16h a  = frag_sw(sA, kt, lane);
      v16h b0 = frag_w(Wh16, 16, kt, nt0, lane);
      v16h b1 = frag_w(Wh16, 16, kt, nt0 + 1, lane);
      acc0 = wmma_f32_16x16x32(a, b0, acc0);
      acc1 = wmma_f32_16x16x32(a, b1, acc1);
    }
    const int nA = n0 + (lane & 15), nB = nA + 16;
    const float bA = Wh_b[nA], bB = Wh_b[nB];
#pragma unroll
    for (int v = 0; v < 8; ++v) {
      const int m = v + ((lane >> 4) << 3);
      const float mk = mask[row0 + m];
      const int   mg = mid[row0 + m];
      {
        const float z = (float)sZ[m * HH + nA];
        const float pre = fast_tanh(acc0[v] + bA);
        h[(size_t)mg * HH + nA] =
            ((1.0f - z) * sSumH[m * HH + nA] + z * pre) * mk;
      }
      {
        const float z = (float)sZ[m * HH + nB];
        const float pre = fast_tanh(acc1[v] + bB);
        h[(size_t)mg * HH + nB] =
            ((1.0f - z) * sSumH[m * HH + nB] + z * pre) * mk;
      }
    }
  }
}

// ---------------- readout: relu([emb[root] | sum h[root_nei]] @ W + b) ------
__global__ __launch_bounds__(256) void readout_kernel(
    const float*    __restrict__ emb,
    const _Float16* __restrict__ W16, const float* __restrict__ W_b,
    const float*    __restrict__ h,
    const int*      __restrict__ root_wid, const int* __restrict__ root_nei,
    float*          __restrict__ out) {
  __shared__ __align__(32) _Float16 sA[16 * HH2];

  const int tid  = threadIdx.x;
  const int lane = tid & 31;
  const int wv   = tid >> 5;
  const int row0 = blockIdx.x * 16;
  const int nt0  = wv * 2;
  const int n0   = wv * 32;

  for (int i = tid; i < 16 * HH; i += 256) {
    const int r = i >> 8, c = i & (HH - 1);
    const int w = root_wid[row0 + r];
    sA[((c >> 5) << 9) + swz_off(r, c & 31)] = (_Float16)emb[w * HH + c];
    float s = 0.0f;
#pragma unroll
    for (int nb = 0; nb < MAXNB; ++nb) {
      const int hidx = root_nei[(row0 + r) * MAXNB + nb];
      s += h[(size_t)hidx * HH + c];
    }
    sA[((8 + (c >> 5)) << 9) + swz_off(r, c & 31)] = (_Float16)s;
  }
  __syncthreads();

  v8f acc0 = {}, acc1 = {};
#pragma unroll
  for (int kt = 0; kt < 16; ++kt) {
    v16h a  = frag_sw(sA, kt, lane);
    v16h b0 = frag_w(W16, 16, kt, nt0, lane);
    v16h b1 = frag_w(W16, 16, kt, nt0 + 1, lane);
    acc0 = wmma_f32_16x16x32(a, b0, acc0);
    acc1 = wmma_f32_16x16x32(a, b1, acc1);
  }
  const int nA = n0 + (lane & 15), nB = nA + 16;
  const float bA = W_b[nA], bB = W_b[nB];
#pragma unroll
  for (int v = 0; v < 8; ++v) {
    const int m = v + ((lane >> 4) << 3);
    out[(size_t)(row0 + m) * HH + nA] = fmaxf(acc0[v] + bA, 0.0f);
    out[(size_t)(row0 + m) * HH + nB] = fmaxf(acc1[v] + bB, 0.0f);
  }
}

// ---------------- host-side orchestration ----------------
extern "C" void kernel_launch(void* const* d_in, const int* in_sizes, int n_in,
                              void* d_out, int out_size, void* d_ws, size_t ws_size,
                              hipStream_t stream) {
  const float* emb   = (const float*)d_in[0];
  const float* Wz_w  = (const float*)d_in[1];
  const float* Wz_b  = (const float*)d_in[2];
  const float* Wr_w  = (const float*)d_in[3];
  const float* Ur_w  = (const float*)d_in[4];
  const float* Ur_b  = (const float*)d_in[5];
  const float* Wh_w  = (const float*)d_in[6];
  const float* Wh_b  = (const float*)d_in[7];
  const float* W_w   = (const float*)d_in[8];
  const float* W_b   = (const float*)d_in[9];
  // d_in[10] = h_init (all zeros) — we keep our own copy in workspace
  const int*   step_wid  = (const int*)d_in[11];
  const int*   step_nei  = (const int*)d_in[12];
  const int*   step_msg  = (const int*)d_in[13];
  const float* step_mask = (const float*)d_in[14];
  const int*   root_wid  = (const int*)d_in[15];
  const int*   root_nei  = (const int*)d_in[16];

  // workspace layout: [ h table (f32) | Wz16 | Wr16 | Ur16 | Wh16 | W16 ]
  char* ws = (char*)d_ws;
  float* h = (float*)ws;
  const size_t hbytes = (size_t)HSLOTS * HH * sizeof(float);
  _Float16* Wz16 = (_Float16*)(ws + ((hbytes + 255) & ~(size_t)255));
  _Float16* Wr16 = Wz16 + HH2 * HH;
  _Float16* Ur16 = Wr16 + HH * HH;
  _Float16* Wh16 = Ur16 + HH * HH;
  _Float16* W16  = Wh16 + HH2 * HH;

  hipMemsetAsync(h, 0, hbytes, stream);

  w_swizzle_kernel<<<(HH2 * HH + 255) / 256, 256, 0, stream>>>(Wz_w, Wz16, HH2, HH);
  w_swizzle_kernel<<<(HH  * HH + 255) / 256, 256, 0, stream>>>(Wr_w, Wr16, HH,  HH);
  w_swizzle_kernel<<<(HH  * HH + 255) / 256, 256, 0, stream>>>(Ur_w, Ur16, HH,  HH);
  w_swizzle_kernel<<<(HH2 * HH + 255) / 256, 256, 0, stream>>>(Wh_w, Wh16, HH2, HH);
  w_swizzle_kernel<<<(HH2 * HH + 255) / 256, 256, 0, stream>>>(W_w,  W16,  HH2, HH);

  for (int t = 0; t < TSTEPS; ++t) {
    gru_step_kernel<<<BKROWS / 16, 256, 0, stream>>>(
        emb, Wz16, Wz_b, Wr16, Ur16, Ur_b, Wh16, Wh_b, h,
        step_wid + (size_t)t * BKROWS,
        step_nei + (size_t)t * BKROWS * MAXNB,
        step_msg + (size_t)t * BKROWS,
        step_mask + (size_t)t * BKROWS);
  }

  readout_kernel<<<BB / 16, 256, 0, stream>>>(emb, W16, W_b, h, root_wid,
                                              root_nei, (float*)d_out);
}
